// CrossAttention_4200478015741
// MI455X (gfx1250) — compile-verified, matrix operand order checked by reading, not run on hardware
//
#include <hip/hip_runtime.h>

// ---------------------------------------------------------------------------
// CrossAttention (reverse-complement attention) for MI455X / gfx1250, wave32.
// GEMMs via v_wmma_f32_16x16x32_bf16; weight tiles staged to LDS with the
// Tensor Data Mover (tensor_load_to_lds, TENSORcnt), double-buffered, with
// compile-time K so the software pipeline has no register copies.
// (1-softmax)/(LK-1) @ V rewritten as (colsum(V) - softmax@V)/(LK-1) so the
// attention core is standard flash attention.  Softmax row-sum is computed by
// an extra WMMA against an all-ones fragment; row-max via v_permlane16_b32.
// ---------------------------------------------------------------------------

#define B_   8
#define L_   1024
#define D1_  768
#define D2_  1024
#define H_   16
#define DKV_ 64
#define M_   (B_ * L_)     // 8192 rows
#define HD_  (H_ * DKV_)   // 1024

// LDS B-tile: 64 rows x 64 bf16, padded by TDM to 144B rows (72 u16) to kill
// bank conflicts on 16-row fragment reads (16 rows * 36 banks mod 64: distinct).
#define BROW_ 72
#define BTILE_ (64 * BROW_)

typedef unsigned short u16;
typedef __attribute__((ext_vector_type(16))) __bf16 v16bf;
typedef __attribute__((ext_vector_type(8)))  float  v8f;
typedef __attribute__((ext_vector_type(4)))  unsigned int v4u;
typedef __attribute__((ext_vector_type(8)))  int v8i;
typedef __attribute__((ext_vector_type(4)))  int v4i;

static __device__ __forceinline__ u16 f2bf(float f) {
    unsigned u = __float_as_uint(f);
    u += 0x7FFFu + ((u >> 16) & 1u);          // round-to-nearest-even
    return (u16)(u >> 16);
}
static __device__ __forceinline__ float bf2f(u16 h) {
    return __uint_as_float(((unsigned)h) << 16);
}

union FragU { v16bf v; uint4 u[2]; };

// 16x32 bf16 WMMA fragment from a K-contiguous row pointer.
//   lanes 0-15  : K = {k0+0..7,  k0+16..23}
//   lanes 16-31 : K = {k0+8..15, k0+24..31}
static __device__ __forceinline__ v16bf load_frag(const u16* row, int k0, int hi) {
    FragU f;
    f.u[0] = *(const uint4*)(row + k0 +      hi * 8);
    f.u[1] = *(const uint4*)(row + k0 + 16 + hi * 8);
    return f.v;
}

#define WMMA_BF16(a, b, c) \
    __builtin_amdgcn_wmma_f32_16x16x32_bf16(false, (a), false, (b), (short)0, (c), false, false)

// max with lane (this ^ mask) inside 16-lane groups via v_permlane16_b32 (VALU,
// keeps the LDS pipe free for the P-tile staging traffic).
static __device__ __forceinline__ float plmax(float v, unsigned s0, unsigned s1) {
    unsigned u = __float_as_uint(v);
    unsigned p = __builtin_amdgcn_permlane16(u, u, s0, s1, false, false);
    return fmaxf(v, __uint_as_float(p));
}
static __device__ __forceinline__ float rowmax16(float v) {
    v = plmax(v, 0x67452301u, 0xEFCDAB89u);   // xor 1
    v = plmax(v, 0x54761032u, 0xDCFE98BAu);   // xor 2
    v = plmax(v, 0x32107654u, 0xBA98FEDCu);   // xor 4
    v = plmax(v, 0xFEDCBA98u, 0x76543210u);   // xor 8
    return v;
}

// ---------------------------------------------------------------------------
// TDM: load a 64(N-rows) x 64(K) bf16 tile of Bt[(n0..n0+63), k..k+63] into
// LDS at lds_off, padding each 128B row with 16B (pad_interval=32DW code 4,
// pad_amount=4DW code 3) -> 144B LDS row stride.
// ---------------------------------------------------------------------------
static __device__ __forceinline__ void tdm_load_btile(const u16* gsrc, unsigned lds_off, int Kd) {
    unsigned long long ga = (unsigned long long)(size_t)gsrc;
    v4u g0;
    g0[0] = 1u;                                   // count=1, no gather
    g0[1] = lds_off;                              // lds_addr (bytes)
    g0[2] = (unsigned)ga;                         // global_addr[31:0]
    g0[3] = (unsigned)((ga >> 32) & 0x01FFFFFFull) | (2u << 30);  // addr[56:32] | type=2
    v8i g1;
    g1[0] = (int)((1u << 16)      // data_size = 2 bytes
                | (1u << 20)      // pad_enable
                | (4u << 22)      // pad_interval: 32 DWORDs (128B)
                | (3u << 25));    // pad_amount : 4 DWORDs (16B)
    g1[1] = (int)(((unsigned)Kd & 0xFFFFu) << 16);            // tensor_dim0[15:0]
    g1[2] = (int)((((unsigned)Kd >> 16) & 0xFFFFu) | (1024u << 16)); // dim0 hi | tensor_dim1
    g1[3] = (int)(64u << 16);                                 // tile_dim0 = 64
    g1[4] = 64;                                               // tile_dim1 = 64
    g1[5] = Kd;                                               // tensor_dim0_stride lo
    g1[6] = 0;
    g1[7] = 0;
    v4i gz = {0, 0, 0, 0};
#if __clang_major__ >= 23
    v8i z8 = {0, 0, 0, 0, 0, 0, 0, 0};
    __builtin_amdgcn_tensor_load_to_lds(g0, g1, gz, gz, z8, 0);
#else
    __builtin_amdgcn_tensor_load_to_lds(g0, g1, gz, gz, 0);
#endif
}

// ---------------------------------------------------------------------------
// One GEMM pass: wave computes a 32(M) x 64(N) tile; acc[2][4] f32.
// B staged via double-buffered TDM (compile-time KD, fully unrolled pipeline:
// even/odd steps alternate prefetch register sets -> no copies).  Each LDS B
// fragment feeds two WMMAs (two M-tiles).
// ---------------------------------------------------------------------------
template <int KD>
static __device__ __forceinline__ void gemm_phase(const u16* __restrict__ A,
                                                  const u16* __restrict__ Bt,
                                                  u16* bsm,
                                                  int m0, int n0, int ln, int hi,
                                                  bool wave0, v8f acc[2][4]) {
    constexpr int T = KD / 64;
    static_assert((T & 1) == 0, "KD must be a multiple of 128");
    const u16* ar0 = A + (size_t)(m0 + ln) * KD;
    const u16* ar1 = A + (size_t)(m0 + 16 + ln) * KD;
    if (wave0) tdm_load_btile(Bt + (size_t)n0 * KD, (unsigned)(size_t)bsm, KD);

    v16bf a[2][2], an[2][2];
    a[0][0] = load_frag(ar0, 0, hi);
    a[0][1] = load_frag(ar0, 32, hi);
    a[1][0] = load_frag(ar1, 0, hi);
    a[1][1] = load_frag(ar1, 32, hi);

    auto step = [&](int t, int buf, v16bf (&ac)[2][2], v16bf (&anx)[2][2]) {
        if (wave0) __builtin_amdgcn_s_wait_tensorcnt(0);   // buffer `buf` ready
        __syncthreads();                                   // visible to all waves
        if (t + 1 < T) {                                   // overlap next TDM
            if (wave0)
                tdm_load_btile(Bt + (size_t)n0 * KD + (t + 1) * 64,
                               (unsigned)(size_t)(bsm + (buf ^ 1) * BTILE_), KD);
            anx[0][0] = load_frag(ar0, (t + 1) * 64, hi);  // prefetch next A frags
            anx[0][1] = load_frag(ar0, (t + 1) * 64 + 32, hi);
            anx[1][0] = load_frag(ar1, (t + 1) * 64, hi);
            anx[1][1] = load_frag(ar1, (t + 1) * 64 + 32, hi);
        }
        const u16* cur = bsm + buf * BTILE_;
#pragma unroll
        for (int ks = 0; ks < 2; ks++) {
#pragma unroll
            for (int nt = 0; nt < 4; nt++) {
                const u16* brow = cur + (nt * 16 + ln) * BROW_ + ks * 32;
                FragU f;
                f.u[0] = *(const uint4*)(brow + hi * 8);
                f.u[1] = *(const uint4*)(brow + 16 + hi * 8);
                acc[0][nt] = WMMA_BF16(ac[0][ks], f.v, acc[0][nt]);
                acc[1][nt] = WMMA_BF16(ac[1][ks], f.v, acc[1][nt]);
            }
        }
    };
#pragma unroll
    for (int t = 0; t < T; t += 2) {
        step(t, 0, a, an);
        step(t + 1, 1, an, a);
    }
}

// ---------------------------------------------------------------------------
// fp32 -> bf16 conversion kernels
// ---------------------------------------------------------------------------
__global__ __launch_bounds__(256) void cvt_kernel(const float* in, u16* out, int n) {
    int i = blockIdx.x * 256 + threadIdx.x;
    if (i < n) out[i] = f2bf(in[i]);
}

// outT[c*R + r] = in[r*C + c]   (weights: [K,N] row-major -> [N,K] row-major)
__global__ __launch_bounds__(256) void cvtT_kernel(const float* in, u16* out, int R, int C) {
    int i = blockIdx.x * 256 + threadIdx.x;
    if (i < R * C) {
        int c = i / R, r = i - c * R;
        out[i] = f2bf(in[r * C + c]);
    }
}

// ---------------------------------------------------------------------------
// Projection GEMM: C[M,N] = scale * A @ B.  Block = 8 waves, tile 256x64.
// mode 0: bf16 row-major [M, 1024];  mode 1: head-transposed Vt for attention.
// ---------------------------------------------------------------------------
template <int KD>
__global__ __launch_bounds__(256) void gemm_bf16_kernel(const u16* __restrict__ A,
                                                        const u16* __restrict__ Bt,
                                                        u16* __restrict__ out,
                                                        float scale, int mode) {
    __shared__ u16 bsm[2 * BTILE_];
    const int wave = threadIdx.x >> 5;
    const int lane = threadIdx.x & 31;
    const int ln = lane & 15, hi = lane >> 4;
    const int m0 = blockIdx.x * 256 + wave * 32;
    const int n0 = blockIdx.y * 64;

    v8f acc[2][4] = {};
    gemm_phase<KD>(A, Bt, bsm, m0, n0, ln, hi, threadIdx.x < 32, acc);

#pragma unroll
    for (int mt = 0; mt < 2; mt++) {
#pragma unroll
        for (int nt = 0; nt < 4; nt++) {
#pragma unroll
            for (int r = 0; r < 8; r++) {
                int row = m0 + mt * 16 + hi * 8 + r;
                int col = n0 + nt * 16 + ln;
                u16 v = f2bf(acc[mt][nt][r] * scale);
                if (mode == 0) {
                    out[(size_t)row * HD_ + col] = v;
                } else {
                    int bb = row >> 10, ll = row & 1023;
                    int hh = col >> 6, dv = col & 63;
                    out[(size_t)((bb * H_ + hh) * DKV_ + dv) * L_ + ll] = v;
                }
            }
        }
    }
}

// ---------------------------------------------------------------------------
// Vsum[b*H+h][dv] = sum_l V[b,h,l,dv]   (Vt rows contiguous over l)
// ---------------------------------------------------------------------------
__global__ __launch_bounds__(256) void vsum_kernel(const u16* __restrict__ Vt,
                                                   float* __restrict__ Vsum) {
    int i = blockIdx.x * 256 + threadIdx.x;
    const u16* row = Vt + (size_t)i * L_;
    float s = 0.f;
    for (int l = 0; l < L_; l += 8) {
        uint4 c = *(const uint4*)(row + l);
        const u16* p = (const u16*)&c;
#pragma unroll
        for (int j = 0; j < 8; j++) s += bf2f(p[j]);
    }
    Vsum[i] = s;
}

// ---------------------------------------------------------------------------
// Flash attention with RCA fixup.  One wave = one (b, h, 16-query tile).
// Row-sum of P accumulated by WMMA against an all-ones fragment (inherits the
// alpha rescale like O); row-max via v_permlane16_b32 butterflies.
// ---------------------------------------------------------------------------
__global__ __launch_bounds__(128) void attn_kernel(const u16* __restrict__ Qh,
                                                   const u16* __restrict__ Kh,
                                                   const u16* __restrict__ Vt,
                                                   const float* __restrict__ Vsum,
                                                   u16* __restrict__ Ob) {
    __shared__ u16 plds[4][16 * 32];          // per-wave 16x32 bf16 P staging
    const int wave = threadIdx.x >> 5;
    const int lane = threadIdx.x & 31;
    const int ln = lane & 15, hi = lane >> 4;
    const int w = blockIdx.x * 4 + wave;      // 0 .. 8191
    const int qt = w & 63, h = (w >> 6) & 15, b = w >> 10;

    const u16* qrow = Qh + (size_t)((b << 10) + (qt << 4) + ln) * HD_ + (h << 6);
    const v16bf aq0 = load_frag(qrow, 0, hi);
    const v16bf aq1 = load_frag(qrow, 32, hi);

    FragU ones;                               // bf16 1.0 in every element
    ones.u[0] = make_uint4(0x3F803F80u, 0x3F803F80u, 0x3F803F80u, 0x3F803F80u);
    ones.u[1] = ones.u[0];

    v8f o[4] = {};
    v8f la = {};                              // row-sum accumulator (P @ ones)
    float m[8];
#pragma unroll
    for (int r = 0; r < 8; r++) m[r] = -1e30f;

    for (int kt = 0; kt < L_; kt += 32) {
        // issue V fragment loads early: latency hides under the softmax VALU
        v16bf bv[4];
#pragma unroll
        for (int nt = 0; nt < 4; nt++) {
            const u16* vrow = Vt + (size_t)(((b * H_ + h) << 6) + (nt << 4) + ln) * L_ + kt;
            bv[nt] = load_frag(vrow, 0, hi);
        }
        float s[2][8];
#pragma unroll
        for (int sub = 0; sub < 2; sub++) {
            const u16* krow = Kh + (size_t)((b << 10) + kt + (sub << 4) + ln) * HD_ + (h << 6);
            v16bf bk0 = load_frag(krow, 0, hi);
            v16bf bk1 = load_frag(krow, 32, hi);
            v8f c = {};
            c = WMMA_BF16(aq0, bk0, c);
            c = WMMA_BF16(aq1, bk1, c);
#pragma unroll
            for (int r = 0; r < 8; r++) s[sub][r] = c[r];
        }
        // online softmax per row (row = hi*8 + r, replicated across 16 lanes)
#pragma unroll
        for (int r = 0; r < 8; r++) {
            float mx = rowmax16(fmaxf(s[0][r], s[1][r]));
            float mn = fmaxf(m[r], mx);
            float alpha = __expf(m[r] - mn);
            m[r] = mn;
            float p0 = __expf(s[0][r] - mn);
            float p1 = __expf(s[1][r] - mn);
            la[r] *= alpha;
#pragma unroll
            for (int nt = 0; nt < 4; nt++) o[nt][r] *= alpha;
            plds[wave][(hi * 8 + r) * 32 + ln]      = f2bf(p0);
            plds[wave][(hi * 8 + r) * 32 + 16 + ln] = f2bf(p1);
        }
        __builtin_amdgcn_wave_barrier();      // keep ds stores before ds loads
        // P tile as A-fragment (LDS transpose: C-layout -> A-layout)
        FragU pa;
        const u16* prow = &plds[wave][ln * 32];
        pa.u[0] = *(const uint4*)(prow + hi * 8);
        pa.u[1] = *(const uint4*)(prow + 16 + hi * 8);
        la = WMMA_BF16(pa.v, ones.v, la);     // row sums, replicated per column
#pragma unroll
        for (int nt = 0; nt < 4; nt++) o[nt] = WMMA_BF16(pa.v, bv[nt], o[nt]);
    }

    // epilogue: out = (Vsum - O/l) / (LK-1)
    const int brow = (b << 10) + (qt << 4);
    const float invLm1 = 1.0f / (float)(L_ - 1);
#pragma unroll
    for (int nt = 0; nt < 4; nt++) {
#pragma unroll
        for (int r = 0; r < 8; r++) {
            int row = hi * 8 + r;
            int dv = (nt << 4) + ln;
            float vs = Vsum[((b * H_ + h) << 6) + dv];
            float val = (vs - o[nt][r] / la[r]) * invLm1;
            Ob[(size_t)(brow + row) * HD_ + (h << 6) + dv] = f2bf(val);
        }
    }
}

// ---------------------------------------------------------------------------
// Dual GEMM: Y[m,n] = Ob@fc_w + q@resid_w + bias  (f32 out), TDM-staged B.
// ---------------------------------------------------------------------------
__global__ __launch_bounds__(256) void gemm_dual_kernel(const u16* __restrict__ A1,
                                                        const u16* __restrict__ B1t,
                                                        const u16* __restrict__ A2,
                                                        const u16* __restrict__ B2t,
                                                        const float* __restrict__ bias,
                                                        float* __restrict__ Y) {
    __shared__ u16 bsm[2 * BTILE_];
    const int wave = threadIdx.x >> 5;
    const int lane = threadIdx.x & 31;
    const int ln = lane & 15, hi = lane >> 4;
    const int m0 = blockIdx.x * 256 + wave * 32;
    const int n0 = blockIdx.y * 64;

    v8f acc[2][4] = {};
    gemm_phase<D2_>(A1, B1t, bsm, m0, n0, ln, hi, threadIdx.x < 32, acc);
    gemm_phase<D1_>(A2, B2t, bsm, m0, n0, ln, hi, threadIdx.x < 32, acc);

#pragma unroll
    for (int mt = 0; mt < 2; mt++) {
#pragma unroll
        for (int nt = 0; nt < 4; nt++) {
#pragma unroll
            for (int r = 0; r < 8; r++) {
                int row = m0 + mt * 16 + hi * 8 + r;
                int col = n0 + nt * 16 + ln;
                Y[(size_t)row * HD_ + col] = acc[mt][nt][r] + bias[col];
            }
        }
    }
}

// ---------------------------------------------------------------------------
// LayerNorm over last dim (1024), one block per row.
// ---------------------------------------------------------------------------
__global__ __launch_bounds__(256) void ln_kernel(const float* __restrict__ Y,
                                                 const float* __restrict__ gamma,
                                                 const float* __restrict__ beta,
                                                 float* __restrict__ out) {
    __shared__ float r1[256], r2[256];
    const int row = blockIdx.x, tid = threadIdx.x;
    const float* x = Y + ((size_t)row << 10);
    float s = 0.f, s2 = 0.f;
    for (int i = tid; i < D2_; i += 256) { float v = x[i]; s += v; s2 += v * v; }
    r1[tid] = s; r2[tid] = s2;
    __syncthreads();
    for (int st = 128; st > 0; st >>= 1) {
        if (tid < st) { r1[tid] += r1[tid + st]; r2[tid] += r2[tid + st]; }
        __syncthreads();
    }
    float mean = r1[0] * (1.0f / D2_);
    float var  = r2[0] * (1.0f / D2_) - mean * mean;
    float inv  = rsqrtf(var + 1e-5f);
    for (int i = tid; i < D2_; i += 256)
        out[((size_t)row << 10) + i] = (x[i] - mean) * inv * gamma[i] + beta[i];
}

// ---------------------------------------------------------------------------
extern "C" void kernel_launch(void* const* d_in, const int* in_sizes, int n_in,
                              void* d_out, int out_size, void* d_ws, size_t ws_size,
                              hipStream_t stream) {
    const float* q       = (const float*)d_in[0];
    const float* k       = (const float*)d_in[1];
    const float* v       = (const float*)d_in[2];
    const float* w_qs    = (const float*)d_in[3];
    const float* w_ks    = (const float*)d_in[4];
    const float* w_vs    = (const float*)d_in[5];
    const float* fc_w    = (const float*)d_in[6];
    const float* resid_w = (const float*)d_in[7];
    const float* resid_b = (const float*)d_in[8];
    const float* ln_g    = (const float*)d_in[9];
    const float* ln_b    = (const float*)d_in[10];
    float* outp = (float*)d_out;

    char* ws = (char*)d_ws;
    size_t off = 0;
    auto alloc = [&](size_t bytes) {
        void* p = ws + off;
        off += (bytes + 255) & ~(size_t)255;
        return p;
    };
    u16* qb     = (u16*)alloc((size_t)M_ * D1_ * 2);
    u16* kb     = (u16*)alloc((size_t)M_ * D2_ * 2);
    u16* vb     = (u16*)alloc((size_t)M_ * D2_ * 2);
    u16* wqT    = (u16*)alloc((size_t)HD_ * D1_ * 2);
    u16* wkT    = (u16*)alloc((size_t)HD_ * D2_ * 2);
    u16* wvT    = (u16*)alloc((size_t)HD_ * D2_ * 2);
    u16* fcT    = (u16*)alloc((size_t)D2_ * HD_ * 2);
    u16* residT = (u16*)alloc((size_t)HD_ * D1_ * 2);
    u16* Qh     = (u16*)alloc((size_t)M_ * HD_ * 2);
    u16* Kh     = (u16*)alloc((size_t)M_ * HD_ * 2);
    u16* Vt     = (u16*)alloc((size_t)M_ * HD_ * 2);
    float* Vsum = (float*)alloc((size_t)B_ * H_ * DKV_ * 4);
    u16* Ob     = (u16*)alloc((size_t)M_ * HD_ * 2);
    float* Yws  = (float*)alloc((size_t)M_ * HD_ * 4);
    (void)ws_size; (void)in_sizes; (void)n_in; (void)out_size;

    cvt_kernel<<<(M_ * D1_) / 256, 256, 0, stream>>>(q, qb, M_ * D1_);
    cvt_kernel<<<(M_ * D2_) / 256, 256, 0, stream>>>(k, kb, M_ * D2_);
    cvt_kernel<<<(M_ * D2_) / 256, 256, 0, stream>>>(v, vb, M_ * D2_);
    cvtT_kernel<<<(D1_ * HD_) / 256, 256, 0, stream>>>(w_qs, wqT, D1_, HD_);
    cvtT_kernel<<<(D2_ * HD_) / 256, 256, 0, stream>>>(w_ks, wkT, D2_, HD_);
    cvtT_kernel<<<(D2_ * HD_) / 256, 256, 0, stream>>>(w_vs, wvT, D2_, HD_);
    cvtT_kernel<<<(HD_ * D2_) / 256, 256, 0, stream>>>(fc_w, fcT, HD_, D2_);
    cvtT_kernel<<<(D1_ * HD_) / 256, 256, 0, stream>>>(resid_w, residT, D1_, HD_);

    dim3 gg(M_ / 256, HD_ / 64);
    gemm_bf16_kernel<D1_><<<gg, 256, 0, stream>>>(qb, wqT, Qh, 0.125f, 0);
    gemm_bf16_kernel<D2_><<<gg, 256, 0, stream>>>(kb, wkT, Kh, 1.0f, 0);
    gemm_bf16_kernel<D2_><<<gg, 256, 0, stream>>>(vb, wvT, Vt, 1.0f, 1);

    vsum_kernel<<<(B_ * H_ * DKV_) / 256, 256, 0, stream>>>(Vt, Vsum);

    attn_kernel<<<(B_ * H_ * (L_ / 16)) / 4, 128, 0, stream>>>(Qh, Kh, Vt, Vsum, Ob);

    gemm_dual_kernel<<<gg, 256, 0, stream>>>(Ob, fcT, qb, residT, resid_b, Yws);

    ln_kernel<<<M_, 256, 0, stream>>>(Yws, ln_g, ln_b, outp);
}
